// HybridModel_7215545057440
// MI455X (gfx1250) — matrix operand order repaired and checked by reading, not compiled, not run on hardware
//
#include <hip/hip_runtime.h>
#include <math.h>

typedef __attribute__((ext_vector_type(16))) _Float16 v16h;
typedef __attribute__((ext_vector_type(8)))  float    v8f;

// ---------------- compile-time construction of the 16x16 circuit unitary ----------------
struct M16d { double a[16][16]; };

static constexpr M16d matmul16(const M16d& A, const M16d& B) {
  M16d C{};
  for (int i = 0; i < 16; ++i)
    for (int j = 0; j < 16; ++j) {
      double s = 0.0;
      for (int t = 0; t < 16; ++t) s += A.a[i][t] * B.a[t][j];
      C.a[i][j] = s;
    }
  return C;
}

// Embed a 2^k x 2^k gate on `wires` (wire 0 = MSB) into 16x16.
static constexpr M16d embed(const double* mat, int k, const int* wires) {
  M16d full{};
  const int dimk = 1 << k;
  for (int j = 0; j < 16; ++j) {
    int bits[4] = { (j >> 3) & 1, (j >> 2) & 1, (j >> 1) & 1, j & 1 };
    int sub_j = 0;
    for (int idx = 0; idx < k; ++idx) sub_j = sub_j * 2 + bits[wires[idx]];
    for (int sub_i = 0; sub_i < dimk; ++sub_i) {
      int nb[4] = { bits[0], bits[1], bits[2], bits[3] };
      for (int idx = 0; idx < k; ++idx) nb[wires[idx]] = (sub_i >> (k - 1 - idx)) & 1;
      const int i = (nb[0] << 3) | (nb[1] << 2) | (nb[2] << 1) | nb[3];
      full.a[i][j] = mat[sub_i * dimk + sub_j];
    }
  }
  return full;
}

static constexpr M16d build_U() {
  constexpr double s = 0.70710678118654752440;
  const double H[4]   = { s, s, s, -s };
  const double SW[16] = { 1,0,0,0, 0,0,1,0, 0,1,0,0, 0,0,0,1 };
  const double CH[16] = { 1,0,0,0, 0,s,0,s, 0,0,1,0, 0,s,0,-s };
  double CSW[64] = {};
  for (int i = 0; i < 8; ++i) CSW[i * 8 + i] = 1.0;
  CSW[5 * 8 + 5] = 0.0; CSW[6 * 8 + 6] = 0.0;
  CSW[5 * 8 + 6] = 1.0; CSW[6 * 8 + 5] = 1.0;
  M16d U{};
  for (int i = 0; i < 16; ++i) U.a[i][i] = 1.0;
  { const int w[1] = {3};       U = matmul16(embed(H,  1, w), U); }
  { const int w[2] = {3,2};     U = matmul16(embed(SW, 2, w), U); }
  { const int w[2] = {2,1};     U = matmul16(embed(SW, 2, w), U); }
  { const int w[2] = {1,0};     U = matmul16(embed(SW, 2, w), U); }
  { const int w[2] = {2,3};     U = matmul16(embed(CH, 2, w), U); }
  { const int w[3] = {3,2,1};   U = matmul16(embed(CSW,3, w), U); }
  { const int w[3] = {3,1,0};   U = matmul16(embed(CSW,3, w), U); }
  return U;
}

struct UF { float a[16][16]; };
static constexpr UF make_Uf() {
  M16d U = build_U();
  UF r{};
  for (int i = 0; i < 16; ++i)
    for (int j = 0; j < 16; ++j) r.a[i][j] = (float)U.a[i][j];
  return r;
}
__device__ constexpr UF d_U = make_Uf();

// ---------------- kernels ----------------

__global__ void zero_ws_kernel(float* ws) { ws[threadIdx.x] = 0.0f; }

// Fused: patch angles -> product state -> U apply (WMMA, f16 hi/lo split) ->
// probs -> sign-combined classifier partial dot -> atomic logits accumulation.
// grid = (16, B), block = 256 (8 waves). Each block: 64 WMMA tiles (16 patches each).
__global__ __launch_bounds__(256) void quanv_wmma_kernel(
    const float* __restrict__ x, const float* __restrict__ W,
    float* __restrict__ logits_acc)
{
  const int lane = threadIdx.x & 31;
  const int wv   = threadIdx.x >> 5;     // wave in block: 0..7
  const int n    = lane & 15;            // N column / A-row patch slot
  const int h    = lane >> 4;            // lane-half selector
  const int img  = blockIdx.y;
  const int tStart = blockIdx.x * 64;    // 1024 tiles per image, 64 per block

  const float* xim = x + (size_t)img * 65536;   // [256][256]

  // Constant B = U^T (32x16, K=16..31 zero-padded), f16 hi/lo split.
  // Dense 16-bit B layout: lane half h holds K = e + 8*h for e=0..7,
  // elements 8..15 hold K>=16 (all zero here).
  v16h bh = {}, bl = {};
  {
#pragma unroll
    for (int e = 0; e < 8; ++e) {
      const float bv = d_U.a[n][8 * h + e];   // B[k=8h+e][n] = U[n][8h+e]
      const _Float16 hi = (_Float16)bv;
      bh[e] = hi;
      bl[e] = (_Float16)(bv - (float)hi);
    }
  }

  // PauliZ signs for this lane's state index n (wire 0 = MSB)
  const float sg0 = 1.f - 2.f * (float)((n >> 3) & 1);
  const float sg1 = 1.f - 2.f * (float)((n >> 2) & 1);
  const float sg2 = 1.f - 2.f * (float)((n >> 1) & 1);
  const float sg3 = 1.f - 2.f * (float)( n       & 1);

  float acc0 = 0.f, acc1 = 0.f;

  for (int i = 0; i < 8; ++i) {
    const int t = tStart + i * 8 + wv;        // tile = 16 consecutive patches in a row

    // ---- build A: state amplitudes of patch p = 16*t + n ----
    const int pA  = 16 * t + n;
    const int prA = pA >> 7;                  // patch row
    const int pcA = pA & 127;                 // patch col
    const float2 r0 = *(const float2*)(xim + (2 * prA    ) * 256 + 2 * pcA);
    const float2 r1 = *(const float2*)(xim + (2 * prA + 1) * 256 + 2 * pcA);
    float c0, s0, c1q, s1q, c2q, s2q, c3q, s3q;
    sincosf(0.5f * r0.x, &s0,  &c0);          // wire 0
    sincosf(0.5f * r0.y, &s1q, &c1q);         // wire 1
    sincosf(0.5f * r1.x, &s2q, &c2q);         // wire 2
    sincosf(0.5f * r1.y, &s3q, &c3q);         // wire 3

    // 16-bit A 16x32 layout: lane holds K = e + 8*h for e=0..7; K>=16 zero-padded.
    // state s = 8a+4b+2c+d with a = h fixed per lane-half.
    const float va = h ? s0 : c0;
    v16h ah = {}, al = {};
#pragma unroll
    for (int e = 0; e < 8; ++e) {
      const float vb = (e & 4) ? s1q : c1q;
      const float vc = (e & 2) ? s2q : c2q;
      const float vd = (e & 1) ? s3q : c3q;
      const float st = va * vb * vc * vd;
      const _Float16 hi = (_Float16)st;
      ah[e] = hi;
      al[e] = (_Float16)(st - (float)hi);
    }

    // ---- OUT(16x16) = STATE x U^T, f16-split (error ~2^-22), f32 accumulate ----
    v8f d = {};
    d = __builtin_amdgcn_wmma_f32_16x16x32_f16(false, al, false, bh, (short)0, d, false, false);
    d = __builtin_amdgcn_wmma_f32_16x16x32_f16(false, ah, false, bl, (short)0, d, false, false);
    d = __builtin_amdgcn_wmma_f32_16x16x32_f16(false, ah, false, bh, (short)0, d, false, false);

    // ---- probs -> fold PauliZ signs directly into classifier weights ----
    // D layout: d[r] = OUT[m][n] with m = r + 8*h. feats index = 4*patch + w.
#pragma unroll
    for (int r = 0; r < 8; ++r) {
      const float amp = d[r];
      const float p   = amp * amp;
      const int  m     = r + 8 * h;
      const int  fbase = (16 * t + m) * 4;
      const float4 w0 = *(const float4*)(W + fbase);
      const float4 w1 = *(const float4*)(W + 65536 + fbase);
      acc0 += p * (sg0 * w0.x + sg1 * w0.y + sg2 * w0.z + sg3 * w0.w);
      acc1 += p * (sg0 * w1.x + sg1 * w1.y + sg2 * w1.z + sg3 * w1.w);
    }
  }

  // wave32 reduction (sums over n and both lane-halves), then one atomic per wave
#pragma unroll
  for (int off = 16; off > 0; off >>= 1) {
    acc0 += __shfl_xor(acc0, off, 32);
    acc1 += __shfl_xor(acc1, off, 32);
  }
  if (lane == 0) {
    atomicAdd(&logits_acc[2 * img + 0], acc0);
    atomicAdd(&logits_acc[2 * img + 1], acc1);
  }
}

__global__ void finalize_kernel(const float* __restrict__ ws,
                                const float* __restrict__ bias,
                                float* __restrict__ out, int B)
{
  const int b = blockIdx.x * blockDim.x + threadIdx.x;
  if (b < B) {
    const float l0 = ws[2 * b + 0] + bias[0];
    const float l1 = ws[2 * b + 1] + bias[1];
    const float mx = fmaxf(l0, l1);
    const float e0 = expf(l0 - mx);
    const float e1 = expf(l1 - mx);
    const float inv = 1.0f / (e0 + e1);
    out[2 * b + 0] = e0 * inv;
    out[2 * b + 1] = e1 * inv;
  }
}

extern "C" void kernel_launch(void* const* d_in, const int* in_sizes, int n_in,
                              void* d_out, int out_size, void* d_ws, size_t ws_size,
                              hipStream_t stream) {
  const float* x    = (const float*)d_in[0];   // [B,256,256]
  const float* W    = (const float*)d_in[1];   // [2,65536]
  const float* bias = (const float*)d_in[2];   // [2]
  float* out = (float*)d_out;                  // [B,2]
  float* ws  = (float*)d_ws;                   // logits accumulators, 2*B floats

  const int B = in_sizes[0] / 65536;           // 128

  zero_ws_kernel<<<1, 2 * B, 0, stream>>>(ws);
  dim3 grid(16, B);
  quanv_wmma_kernel<<<grid, 256, 0, stream>>>(x, W, ws);
  finalize_kernel<<<1, B, 0, stream>>>(ws, bias, out, B);
}
